// MultiHeadSelfAttention_46179488366625
// MI455X (gfx1250) — matrix operand order repaired and checked by reading, not compiled, not run on hardware
//
#include <hip/hip_runtime.h>
#include <hip/hip_bf16.h>

// ---------------------------------------------------------------------------
// Problem constants (from reference): S=2048, B=4, D=1024, H=16, HD=64
// ---------------------------------------------------------------------------
#define SS 2048
#define BB 4
#define DD 1024
#define HH 16
#define HDD 64
#define NEGBIG (-1.0e9f)

typedef __attribute__((ext_vector_type(16))) __bf16 bf16x16;
typedef __attribute__((ext_vector_type(8)))  __bf16 bf16x8;
typedef __attribute__((ext_vector_type(8)))  float  f32x8;

// ---------------------------------------------------------------------------
// WMMA helpers (CDNA5: V_WMMA_F32_16X16X32_BF16, D = A(16x32) * B(32x16) + C)
// ---------------------------------------------------------------------------
__device__ __forceinline__ f32x8 wmma_bf16(bf16x16 a, bf16x16 b, f32x8 c) {
  // 8 args: (neg_a, A, neg_b, B, c_mod, C, reuse_a, reuse_b)
  return __builtin_amdgcn_wmma_f32_16x16x32_bf16(false, a, false, b,
                                                 (short)0, c, false, false);
}

__device__ __forceinline__ bf16x16 concat8(bf16x8 lo, bf16x8 hi) {
  bf16x16 r;
#pragma unroll
  for (int i = 0; i < 8; ++i) { r[i] = lo[i]; r[i + 8] = hi[i]; }
  return r;
}

// A-fragment (16x32, row-major source, row stride ld elements):
// lane<16: M=lane, K {0..7,16..23}; lane>=16: M=lane-16, K {8..15,24..31}
__device__ __forceinline__ bf16x16 load_afrag(const __bf16* p, int ld, int lane) {
  int m    = lane & 15;
  int koff = (lane >> 4) << 3;   // 0 or 8
  const __bf16* r = p + (size_t)m * ld + koff;
  bf16x8 lo = *(const bf16x8*)(r);
  bf16x8 hi = *(const bf16x8*)(r + 16);
  return concat8(lo, hi);
}

// B-fragment (32x16), source stored N-major (i.e. B^T row-major, row stride ld):
// lane<16: N=lane, K {0..7,16..23}; lane>=16: N=lane-16, K {8..15,24..31}
__device__ __forceinline__ bf16x16 load_bfrag(const __bf16* p, int ld, int lane) {
  int n    = lane & 15;
  int koff = (lane >> 4) << 3;   // 0 or 8
  const __bf16* r = p + (size_t)n * ld + koff;
  bf16x8 lo = *(const bf16x8*)(r);
  bf16x8 hi = *(const bf16x8*)(r + 16);
  return concat8(lo, hi);
}

// ---------------------------------------------------------------------------
// 1) pack x (S,B,D) f32 -> xb (B,S,D) bf16
// ---------------------------------------------------------------------------
__global__ void k_pack_x(const float* __restrict__ x, __bf16* __restrict__ xb) {
  size_t idx = (size_t)blockIdx.x * blockDim.x + threadIdx.x;
  size_t n = (size_t)SS * BB * DD;
  if (idx >= n) return;
  int d = (int)(idx % DD);
  int b = (int)((idx / DD) % BB);
  int s = (int)(idx / ((size_t)BB * DD));
  xb[((size_t)b * SS + s) * DD + d] = (__bf16)x[idx];
}

// ---------------------------------------------------------------------------
// 2) transpose + cast weights: w (rows,cols) f32 -> wt (cols,rows) bf16
// ---------------------------------------------------------------------------
__global__ void k_transpose_w(const float* __restrict__ w, __bf16* __restrict__ wt,
                              int rows, int cols) {
  size_t idx = (size_t)blockIdx.x * blockDim.x + threadIdx.x;
  size_t n = (size_t)rows * cols;
  if (idx >= n) return;
  int r = (int)(idx % rows);
  int c = (int)(idx / rows);
  wt[idx] = (__bf16)w[(size_t)r * cols + c];
}

// ---------------------------------------------------------------------------
// 3) QKV projection: qkv = xb @ Wqkv  (per batch, WMMA bf16)
//    Epilogue scatters to q (B,H,S,HD), k (B,H,S,HD), vT (B,H,HD,S)
//    Macro tile 32(M)x256(N) per 256-thread WG = 8 waves (2M x 4N), wave tile 16x64
// ---------------------------------------------------------------------------
__global__ __launch_bounds__(256) void k_gemm_qkv(
    const __bf16* __restrict__ xb, const __bf16* __restrict__ wt,
    __bf16* __restrict__ qbuf, __bf16* __restrict__ kbuf,
    __bf16* __restrict__ vtbuf) {
  const int b    = blockIdx.z;
  const int m0   = blockIdx.y * 32;
  const int n0   = blockIdx.x * 256;
  const int w    = threadIdx.x >> 5;
  const int lane = threadIdx.x & 31;
  const int mrow = m0 + (w >> 2) * 16;
  const int ncol = n0 + (w & 3) * 64;

  const __bf16* A = xb + (size_t)b * SS * DD + (size_t)mrow * DD;
  f32x8 c[4] = {};

  for (int k = 0; k < DD; k += 32) {
    bf16x16 a = load_afrag(A + k, DD, lane);
#pragma unroll
    for (int j = 0; j < 4; ++j) {
      bf16x16 bf = load_bfrag(wt + (size_t)(ncol + j * 16) * DD + k, DD, lane);
      c[j] = wmma_bf16(a, bf, c[j]);
    }
  }

  const int nloc = lane & 15;
  const int moff = (lane >> 4) << 3;
#pragma unroll
  for (int j = 0; j < 4; ++j) {
    int e    = ncol + j * 16 + nloc;
    int part = e >> 10;      // 0=q 1=k 2=v
    int d    = e & 1023;
    int h    = d >> 6;
    int hd   = d & 63;
#pragma unroll
    for (int r = 0; r < 8; ++r) {
      int srow = mrow + moff + r;
      __bf16 bv = (__bf16)c[j][r];
      if (part == 0)
        qbuf[(((size_t)b * HH + h) * SS + srow) * HDD + hd] = bv;
      else if (part == 1)
        kbuf[(((size_t)b * HH + h) * SS + srow) * HDD + hd] = bv;
      else
        vtbuf[(((size_t)b * HH + h) * HDD + hd) * SS + srow] = bv;
    }
  }
}

// ---------------------------------------------------------------------------
// 4) RoPE in place on q (fused 1/sqrt(HD)) and k. Interleaved-pair variant:
//    out[2i]   = x[2i]*cos[2i]   - x[2i+1]*sin[2i]
//    out[2i+1] = x[2i+1]*cos[2i+1] + x[2i]*sin[2i+1]
// ---------------------------------------------------------------------------
__global__ void k_rope(__bf16* __restrict__ qbuf, __bf16* __restrict__ kbuf,
                       const float* __restrict__ cosb, const float* __restrict__ sinb) {
  size_t idx = (size_t)blockIdx.x * blockDim.x + threadIdx.x;
  size_t total = (size_t)BB * HH * SS * (HDD / 2);
  if (idx >= total) return;
  int pair   = (int)(idx & (HDD / 2 - 1));
  size_t t   = idx >> 5;           // (b,h,s) flat
  int srow   = (int)(t & (SS - 1));
  int p0 = pair * 2, p1 = p0 + 1;
  float c0 = cosb[(size_t)srow * HDD + p0], c1 = cosb[(size_t)srow * HDD + p1];
  float s0 = sinb[(size_t)srow * HDD + p0], s1 = sinb[(size_t)srow * HDD + p1];
  size_t base = t * HDD;
  float q0 = (float)qbuf[base + p0], q1 = (float)qbuf[base + p1];
  float k0 = (float)kbuf[base + p0], k1 = (float)kbuf[base + p1];
  const float sc = 0.125f;  // 1/sqrt(64)
  qbuf[base + p0] = (__bf16)((q0 * c0 - q1 * s0) * sc);
  qbuf[base + p1] = (__bf16)((q1 * c1 + q0 * s1) * sc);
  kbuf[base + p0] = (__bf16)(k0 * c0 - k1 * s0);
  kbuf[base + p1] = (__bf16)(k1 * c1 + k0 * s1);
}

// ---------------------------------------------------------------------------
// 5) Flash attention: 1 wave per 16-row Q block, 32-key tiles, online softmax.
//    grid = (S/64, B*H), block 128 (4 waves). ctx out: (B,S,D) bf16.
// ---------------------------------------------------------------------------
__global__ __launch_bounds__(128) void k_flash(
    const __bf16* __restrict__ qbuf, const __bf16* __restrict__ kbuf,
    const __bf16* __restrict__ vtbuf, __bf16* __restrict__ ctx) {
  __shared__ __align__(16) __bf16 p_lds[4][16 * 32];

  const int w    = threadIdx.x >> 5;
  const int lane = threadIdx.x & 31;
  const int bh   = blockIdx.y;                 // b*H + h
  const int qb   = blockIdx.x * 4 + w;         // 16-row Q block index
  const int qg0  = qb * 16;

  const __bf16* Q  = qbuf  + (size_t)bh * SS * HDD + (size_t)qg0 * HDD;
  const __bf16* K  = kbuf  + (size_t)bh * SS * HDD;
  const __bf16* VT = vtbuf + (size_t)bh * HDD * SS;

  bf16x16 aq0 = load_afrag(Q, HDD, lane);        // hd 0..31
  bf16x16 aq1 = load_afrag(Q + 32, HDD, lane);   // hd 32..63

  f32x8 o[4] = {};
  float m_i[8], l_i[8];
#pragma unroll
  for (int r = 0; r < 8; ++r) { m_i[r] = -3.0e38f; l_i[r] = 0.0f; }

  const int nkt     = (qg0 + 47) >> 5;           // ceil((qg0+16)/32)
  const int rowbase = qg0 + ((lane >> 4) << 3);

  for (int kt = 0; kt < nkt; ++kt) {
    const int kcol0 = kt * 32;

    // ---- scores S = Q K^T (two 16x16 output fragments over 32 keys) ----
    f32x8 s0 = {}, s1 = {};
    {
      bf16x16 kb00 = load_bfrag(K + (size_t)kcol0 * HDD,            HDD, lane);
      bf16x16 kb01 = load_bfrag(K + (size_t)kcol0 * HDD + 32,       HDD, lane);
      bf16x16 kb10 = load_bfrag(K + (size_t)(kcol0 + 16) * HDD,     HDD, lane);
      bf16x16 kb11 = load_bfrag(K + (size_t)(kcol0 + 16) * HDD + 32, HDD, lane);
      s0 = wmma_bf16(aq0, kb00, s0); s0 = wmma_bf16(aq1, kb01, s0);
      s1 = wmma_bf16(aq0, kb10, s1); s1 = wmma_bf16(aq1, kb11, s1);
    }

    // ---- causal mask (only last tile can cross the diagonal) ----
    if (kt == nkt - 1) {
      int c0 = kcol0 + (lane & 15);
      int c1 = c0 + 16;
#pragma unroll
      for (int r = 0; r < 8; ++r) {
        int row = rowbase + r;
        if (c0 > row) s0[r] += NEGBIG;
        if (c1 > row) s1[r] += NEGBIG;
      }
    }

    // ---- online softmax (row = vgpr r + 8*(lane>=16); reduce across N=16 lanes)
    float pscale[8];
#pragma unroll
    for (int r = 0; r < 8; ++r) {
      float mx = fmaxf(s0[r], s1[r]);
      mx = fmaxf(mx, __shfl_xor(mx, 1, 16));
      mx = fmaxf(mx, __shfl_xor(mx, 2, 16));
      mx = fmaxf(mx, __shfl_xor(mx, 4, 16));
      mx = fmaxf(mx, __shfl_xor(mx, 8, 16));
      float mn = fmaxf(m_i[r], mx);
      pscale[r] = __expf(m_i[r] - mn);
      m_i[r] = mn;
    }
#pragma unroll
    for (int r = 0; r < 8; ++r) {
      float p0 = __expf(s0[r] - m_i[r]);
      float p1 = __expf(s1[r] - m_i[r]);
      s0[r] = p0; s1[r] = p1;
      float sum = p0 + p1;
      sum += __shfl_xor(sum, 1, 16);
      sum += __shfl_xor(sum, 2, 16);
      sum += __shfl_xor(sum, 4, 16);
      sum += __shfl_xor(sum, 8, 16);
      l_i[r] = l_i[r] * pscale[r] + sum;
    }
#pragma unroll
    for (int j = 0; j < 4; ++j)
#pragma unroll
      for (int r = 0; r < 8; ++r) o[j][r] *= pscale[r];

    // ---- relayout P: C-fragment -> LDS -> A-fragment (wave-private tile) ----
    {
      int n  = lane & 15;
      int mo = (lane >> 4) << 3;
      __bf16* pl = &p_lds[w][0];
#pragma unroll
      for (int r = 0; r < 8; ++r) {
        pl[(mo + r) * 32 + n]      = (__bf16)s0[r];
        pl[(mo + r) * 32 + n + 16] = (__bf16)s1[r];
      }
    }
    bf16x16 pa;
    {
      int m  = lane & 15;
      int k0 = (lane >> 4) << 3;
      const __bf16* pl = &p_lds[w][0] + m * 32 + k0;
      bf16x8 lo = *(const bf16x8*)(pl);
      bf16x8 hi = *(const bf16x8*)(pl + 16);
      pa = concat8(lo, hi);
    }

    // ---- O += P @ V (V^T layout -> contiguous B-fragment loads) ----
#pragma unroll
    for (int j = 0; j < 4; ++j) {
      bf16x16 vb = load_bfrag(VT + (size_t)(j * 16) * SS + kcol0, SS, lane);
      o[j] = wmma_bf16(pa, vb, o[j]);
    }
  }

  // ---- normalize and store ctx (B,S,D) bf16 ----
  const int b = bh >> 4;
  const int h = bh & 15;
  float inv[8];
#pragma unroll
  for (int r = 0; r < 8; ++r) inv[r] = 1.0f / l_i[r];
  const int n  = lane & 15;
  const int mo = (lane >> 4) << 3;
#pragma unroll
  for (int j = 0; j < 4; ++j) {
    int hd = j * 16 + n;
#pragma unroll
    for (int r = 0; r < 8; ++r) {
      int srow = qg0 + mo + r;
      ctx[((size_t)b * SS + srow) * DD + h * HDD + hd] = (__bf16)(o[j][r] * inv[r]);
    }
  }
}

// ---------------------------------------------------------------------------
// 6) Output projection: out (S,B,D) f32 = ctx (B,S,D) @ Wout
// ---------------------------------------------------------------------------
__global__ __launch_bounds__(256) void k_gemm_out(
    const __bf16* __restrict__ ctx, const __bf16* __restrict__ wt,
    float* __restrict__ out) {
  const int b    = blockIdx.z;
  const int m0   = blockIdx.y * 32;
  const int n0   = blockIdx.x * 256;
  const int w    = threadIdx.x >> 5;
  const int lane = threadIdx.x & 31;
  const int mrow = m0 + (w >> 2) * 16;
  const int ncol = n0 + (w & 3) * 64;

  const __bf16* A = ctx + (size_t)b * SS * DD + (size_t)mrow * DD;
  f32x8 c[4] = {};

  for (int k = 0; k < DD; k += 32) {
    bf16x16 a = load_afrag(A + k, DD, lane);
#pragma unroll
    for (int j = 0; j < 4; ++j) {
      bf16x16 bf = load_bfrag(wt + (size_t)(ncol + j * 16) * DD + k, DD, lane);
      c[j] = wmma_bf16(a, bf, c[j]);
    }
  }

  const int nloc = lane & 15;
  const int moff = (lane >> 4) << 3;
#pragma unroll
  for (int j = 0; j < 4; ++j) {
    int e = ncol + j * 16 + nloc;
#pragma unroll
    for (int r = 0; r < 8; ++r) {
      int srow = mrow + moff + r;
      out[((size_t)srow * BB + b) * DD + e] = c[j][r];
    }
  }
}

// ---------------------------------------------------------------------------
// Launch
// ---------------------------------------------------------------------------
extern "C" void kernel_launch(void* const* d_in, const int* in_sizes, int n_in,
                              void* d_out, int out_size, void* d_ws, size_t ws_size,
                              hipStream_t stream) {
  (void)in_sizes; (void)n_in; (void)out_size; (void)ws_size;

  const float* x    = (const float*)d_in[0];
  // d_in[1] = attn_mask (causal handled analytically)
  const float* cosb = (const float*)d_in[2];
  const float* sinb = (const float*)d_in[3];
  const float* Wqkv = (const float*)d_in[4];
  const float* Wout = (const float*)d_in[5];
  float* out = (float*)d_out;

  // Workspace layout (bytes)
  char* ws = (char*)d_ws;
  const size_t sz_xb  = (size_t)BB * SS * DD * 2;        // 16 MB
  const size_t sz_wq  = (size_t)3 * DD * DD * 2;         //  6 MB
  const size_t sz_wo  = (size_t)DD * DD * 2;             //  2 MB
  const size_t sz_qkv = (size_t)BB * SS * DD * 2;        // 16 MB each
  __bf16* xb    = (__bf16*)(ws);
  __bf16* wqkvT = (__bf16*)(ws + sz_xb);
  __bf16* woutT = (__bf16*)(ws + sz_xb + sz_wq);
  __bf16* qbuf  = (__bf16*)(ws + sz_xb + sz_wq + sz_wo);
  __bf16* kbuf  = (__bf16*)(ws + sz_xb + sz_wq + sz_wo + sz_qkv);
  __bf16* vtbuf = (__bf16*)(ws + sz_xb + sz_wq + sz_wo + 2 * sz_qkv);
  __bf16* ctx   = (__bf16*)(ws + sz_xb + sz_wq + sz_wo + 3 * sz_qkv);

  // 1) pack x -> bf16 (B,S,D)
  {
    size_t n = (size_t)SS * BB * DD;
    k_pack_x<<<dim3((unsigned)((n + 255) / 256)), dim3(256), 0, stream>>>(x, xb);
  }
  // 2) weight transposes
  {
    size_t n = (size_t)DD * 3 * DD;
    k_transpose_w<<<dim3((unsigned)((n + 255) / 256)), dim3(256), 0, stream>>>(
        Wqkv, wqkvT, DD, 3 * DD);
    size_t m = (size_t)DD * DD;
    k_transpose_w<<<dim3((unsigned)((m + 255) / 256)), dim3(256), 0, stream>>>(
        Wout, woutT, DD, DD);
  }
  // 3) QKV projection (WMMA)
  k_gemm_qkv<<<dim3(3 * DD / 256, SS / 32, BB), dim3(256), 0, stream>>>(
      xb, wqkvT, qbuf, kbuf, vtbuf);
  // 4) RoPE (+ q scale)
  {
    size_t n = (size_t)BB * HH * SS * (HDD / 2);
    k_rope<<<dim3((unsigned)((n + 255) / 256)), dim3(256), 0, stream>>>(
        qbuf, kbuf, cosb, sinb);
  }
  // 5) Flash attention (WMMA)
  k_flash<<<dim3(SS / 64, BB * HH), dim3(128), 0, stream>>>(qbuf, kbuf, vtbuf, ctx);
  // 6) Output projection (WMMA)
  k_gemm_out<<<dim3(DD / 256, SS / 32, BB), dim3(256), 0, stream>>>(ctx, woutT, out);
}
